// PeriodicSetTransformerEncoder_51273319579856
// MI455X (gfx1250) — matrix operand order repaired and checked by reading, not compiled, not run on hardware
//
#include <hip/hip_runtime.h>
#include <hip/hip_bf16.h>
#include <math.h>

// ---------------------------------------------------------------------------
// PeriodicSetTransformerEncoder for MI455X (gfx1250, wave32, WMMA).
// Compute-bound (~206 GFLOP vs ~70MB mandatory HBM traffic @23.3TB/s): all
// GEMMs run as bf16 v_wmma_f32_16x16x32_bf16 with f32 accumulation.
// A-tiles staged to LDS with GLOBAL_LOAD_ASYNC_TO_LDS_B128 (ASYNCcnt path);
// each wave computes a 64x16 strip so one B-fragment feeds 4 WMMAs, and all
// fragment loads are hoisted ahead of the WMMA group for back-to-back issue.
// Workspace use: ~440 MB of d_ws for bf16 weights + intermediates.
// ---------------------------------------------------------------------------

typedef __bf16 bf16;
typedef __attribute__((ext_vector_type(16))) __bf16 v16bf;
typedef __attribute__((ext_vector_type(8)))  __bf16 v8bf;
typedef __attribute__((ext_vector_type(8)))  float  v8f;

#define E_DIM 128
#define H_DIM 4
#define D_DIM 512
#define B_DIM 512
#define N_DIM 128

static __device__ inline v8f vzero8() {
  v8f z;
#pragma unroll
  for (int i = 0; i < 8; ++i) z[i] = 0.f;
  return z;
}

// --- CDNA5 async global->LDS staging (ISA 15.18.3, ASYNCcnt) ---------------
static __device__ inline void async_ld_b128(unsigned lds_off, const void* g) {
  asm volatile("global_load_async_to_lds_b128 %0, %1, off"
               :
               : "v"(lds_off), "v"((unsigned long long)(size_t)g)
               : "memory");
}
static __device__ inline void wait_async0() {
  asm volatile("s_wait_asynccnt 0" ::: "memory");
}

// reductions across a 16-lane half (rows of a WMMA tile live in one half)
static __device__ inline float redmax16(float v) {
#pragma unroll
  for (int m = 1; m <= 8; m <<= 1) v = fmaxf(v, __shfl_xor(v, m, 32));
  return v;
}
static __device__ inline float redsum16(float v) {
#pragma unroll
  for (int m = 1; m <= 8; m <<= 1) v += __shfl_xor(v, m, 32);
  return v;
}
static __device__ inline float redsum32(float v) {
#pragma unroll
  for (int m = 1; m <= 16; m <<= 1) v += __shfl_xor(v, m, 32);
  return v;
}

// --- WMMA fragment loaders (wave32; contiguous k order, same for A and B) --------------
// A (16x32 bf16): lane L holds row M=L%16; K = e + 16*(L/16)
static __device__ inline v16bf load_a_frag(const bf16* base, int ld, int lane) {
  const bf16* p = base + (size_t)(lane & 15) * ld + ((lane >> 4) << 4);
  v8bf lo = *(const v8bf*)(p);
  v8bf hi = *(const v8bf*)(p + 8);
  v16bf a;
#pragma unroll
  for (int e = 0; e < 8; ++e) { a[e] = lo[e]; a[e + 8] = hi[e]; }
  return a;
}
// B (32x16 bf16) sourced from row-major W (N rows x K): B[k][n] = W[n][k].
// lane L holds col N=L%16 (a W row), K = e + 16*(L/16): one 32B vector load.
static __device__ inline v16bf load_b_frag(const bf16* base, int ld, int lane) {
  return *(const v16bf*)(base + (size_t)(lane & 15) * ld + ((lane >> 4) << 4));
}

static __device__ inline v8f wmma_bf16(v16bf a, v16bf b, v8f c) {
  return __builtin_amdgcn_wmma_f32_16x16x32_bf16(false, a, false, b,
                                                 (short)0, c, false, false);
}

// ---------------------------------------------------------------------------
// f32 -> bf16 conversion
__global__ void cvt_f32_bf16(const float* __restrict__ in,
                             bf16* __restrict__ out, long n) {
  for (long i = (long)blockIdx.x * blockDim.x + threadIdx.x; i < n;
       i += (long)gridDim.x * blockDim.x)
    out[i] = (bf16)in[i];
}

// ---------------------------------------------------------------------------
// Generic Y = X * W^T (+bias) GEMM. Block = 8 waves; block tile 64 x 128.
// 64 A-rows are async-staged once into LDS; each wave computes a 64x16 strip
// (4 accumulators). Per k-step: clause 8 ds_loads + 2 global loads, one wait,
// then 4 back-to-back WMMAs sharing the B fragment.
// modes: 0 = bias + bf16 store
//        1 = bias + bf16 store transposed per 128-row batch (for V^T)
//        2 = softplus(acc) + residual -> bf16   (att_output -> h epilogue)
//        3 = bias + f32 store                    (final output)
// wbs: per-128-row batch stride of W in elements (0 = shared weights)
__global__ __launch_bounds__(256, 2)
void gemm_bf16(const bf16* __restrict__ A, const bf16* __restrict__ W,
               const float* __restrict__ bias, const bf16* __restrict__ resid,
               bf16* __restrict__ Yb, float* __restrict__ Yf,
               int M, int N, int K, long wbs, int mode) {
  extern __shared__ char smem_raw[];
  bf16* sA = (bf16*)smem_raw;                       // 64 x K bf16
  const int tile_m = blockIdx.y << 6;
  const int wave = threadIdx.x >> 5;
  const int lane = threadIdx.x & 31;
  const int n0 = (blockIdx.x << 7) + (wave << 4);

  {  // async-stage 64 A-rows (64*K*2 bytes) into LDS, 16B per op
    const char* g = (const char*)(A + (size_t)tile_m * K);
    const int nbytes = K << 7;
    for (int i = threadIdx.x * 16; i < nbytes; i += 256 * 16)
      async_ld_b128((unsigned)(size_t)(smem_raw + i), g + i);
    wait_async0();
  }
  __syncthreads();

  const bf16* Wp = W + (size_t)(tile_m >> 7) * wbs + (size_t)n0 * K;
  v8f acc[4];
#pragma unroll
  for (int t = 0; t < 4; ++t) acc[t] = vzero8();

  for (int kb = 0; kb < K; kb += 32) {
    __builtin_prefetch(Wp + kb + 256, 0, 1);        // global_prefetch_b8
    // hoist all fragment loads -> one wait, 4 back-to-back WMMAs
    v16bf b  = load_b_frag(Wp + kb, K, lane);
    v16bf a0 = load_a_frag(sA + kb, K, lane);
    v16bf a1 = load_a_frag(sA + (size_t)(16) * K + kb, K, lane);
    v16bf a2 = load_a_frag(sA + (size_t)(32) * K + kb, K, lane);
    v16bf a3 = load_a_frag(sA + (size_t)(48) * K + kb, K, lane);
    acc[0] = wmma_bf16(a0, b, acc[0]);
    acc[1] = wmma_bf16(a1, b, acc[1]);
    acc[2] = wmma_bf16(a2, b, acc[2]);
    acc[3] = wmma_bf16(a3, b, acc[3]);
  }

  const int col = n0 + (lane & 15);
  const int rb = (lane >> 4) << 3;                  // D-frag row offset
  const float bv = bias ? bias[col] : 0.f;
#pragma unroll
  for (int t = 0; t < 4; ++t) {
#pragma unroll
    for (int r = 0; r < 8; ++r) {
      const int m = tile_m + (t << 4) + rb + r;
      float v = acc[t][r] + bv;
      if (mode == 2) {
        const float sp = v > 20.f ? v : log1pf(__expf(v));
        v = (float)resid[(size_t)m * N + col] + sp;
      }
      if (mode == 3)
        Yf[(size_t)m * N + col] = v;
      else if (mode == 1)  // Y[b][col][m%128] : per-batch transposed store
        Yb[(size_t)(m >> 7) * ((size_t)N << 7) + ((size_t)col << 7) +
           (m & 127)] = (bf16)v;
      else
        Yb[(size_t)m * N + col] = (bf16)v;
    }
  }
}

// ---------------------------------------------------------------------------
// pad[row] = (sum |xe[row,:]| == 0)
__global__ __launch_bounds__(256)
void pad_kernel(const bf16* __restrict__ xe, float* __restrict__ pad) {
  const int row = (blockIdx.x << 3) + (threadIdx.x >> 5);
  const int lane = threadIdx.x & 31;
  const bf16* p = xe + (size_t)row * D_DIM;
  float s = 0.f;
  for (int i = lane; i < D_DIM; i += 32) s += fabsf((float)p[i]);
  s = redsum32(s);
  if (lane == 0) pad[row] = (s == 0.f) ? 1.f : 0.f;
}

// ---------------------------------------------------------------------------
// Fused attention for one batch per block: scores = q_h k_h^T/sqrt(hd) with
// pad mask, softmax(j), mean over 4 heads, * weights_j, row-renormalize.
// LDS = exactly 64KB (q-head + k-head tiles, async-staged). Softmax / mean /
// renorm fully in registers via 16-lane shuffle reductions (wave32).
__global__ __launch_bounds__(256, 1)
void attn_fused(const bf16* __restrict__ Q, const bf16* __restrict__ Kmat,
                const float* __restrict__ pad, const float* __restrict__ wts,
                bf16* __restrict__ attw) {
  __shared__ bf16 sQ[N_DIM * E_DIM];                // 32 KB
  __shared__ bf16 sK[N_DIM * E_DIM];                // 32 KB
  const int b = blockIdx.x;
  const int wave = threadIdx.x >> 5;
  const int lane = threadIdx.x & 31;
  const int l15 = lane & 15;
  const int half = lane >> 4;

  float padc[8], wcol[8];
#pragma unroll
  for (int tn = 0; tn < 8; ++tn) {
    padc[tn] = pad[(size_t)b * N_DIM + tn * 16 + l15];
    wcol[tn] = wts[(size_t)b * N_DIM + tn * 16 + l15];
  }

  v8f accM[8];                                      // head-mean attn rows
#pragma unroll
  for (int tn = 0; tn < 8; ++tn) accM[tn] = vzero8();

  const float scale = 0.088388347648318447f;        // 1/sqrt(128)

  for (int h = 0; h < H_DIM; ++h) {
    const int coff = h * E_DIM;
    for (int i = threadIdx.x; i < N_DIM * 16; i += 256) {  // 16B chunks
      const int row = i >> 4, c = (i & 15) << 4;
      const size_t gro = ((size_t)b * N_DIM + row) * D_DIM + coff;
      async_ld_b128((unsigned)(size_t)((char*)(sQ + row * E_DIM) + c),
                    (const char*)(Q + gro) + c);
      async_ld_b128((unsigned)(size_t)((char*)(sK + row * E_DIM) + c),
                    (const char*)(Kmat + gro) + c);
    }
    wait_async0();
    __syncthreads();

    // wave owns score rows [wave*16, wave*16+16): one A-frag feeds 8 WMMAs;
    // hoist the 8 B-fragment ds_loads ahead of the WMMA group.
    v8f accS[8];
#pragma unroll
    for (int tn = 0; tn < 8; ++tn) accS[tn] = vzero8();
#pragma unroll
    for (int kb = 0; kb < E_DIM; kb += 32) {
      v16bf a = load_a_frag(sQ + (size_t)(wave << 4) * E_DIM + kb, E_DIM, lane);
      v16bf bb[8];
#pragma unroll
      for (int tn = 0; tn < 8; ++tn)
        bb[tn] = load_b_frag(sK + (size_t)(tn << 4) * E_DIM + kb, E_DIM, lane);
#pragma unroll
      for (int tn = 0; tn < 8; ++tn) accS[tn] = wmma_bf16(a, bb[tn], accS[tn]);
    }

    // softmax over each row (row = wave*16 + r + 8*half; its 128 cols live in
    // accS[0..7][r] across the 16 lanes of this half), accumulate /H mean.
#pragma unroll
    for (int r = 0; r < 8; ++r) {
      float sc[8];
      float mx = -__builtin_inff();
#pragma unroll
      for (int tn = 0; tn < 8; ++tn) {
        float s = (padc[tn] != 0.f) ? -__builtin_inff() : accS[tn][r] * scale;
        sc[tn] = s;
        mx = fmaxf(mx, s);
      }
      mx = redmax16(mx);
      const bool dead = (mx == -__builtin_inff());
      float ex[8];
      float sum = 0.f;
#pragma unroll
      for (int tn = 0; tn < 8; ++tn) {
        const float e = dead ? 0.f : __expf(sc[tn] - mx);
        ex[tn] = e;
        sum += e;
      }
      sum = redsum16(sum);
      const float inv = sum > 0.f ? 0.25f / sum : 0.f;
#pragma unroll
      for (int tn = 0; tn < 8; ++tn) accM[tn][r] += ex[tn] * inv;
    }
    __syncthreads();
  }

  // * weights_j, renormalize rows, store bf16 attention matrix
#pragma unroll
  for (int r = 0; r < 8; ++r) {
    float vv[8];
    float s = 0.f;
#pragma unroll
    for (int tn = 0; tn < 8; ++tn) { vv[tn] = accM[tn][r] * wcol[tn]; s += vv[tn]; }
    s = redsum16(s);
    const float inv = (s != 0.f) ? 1.f / s : 0.f;
    const int row = (wave << 4) + r + (half << 3);
#pragma unroll
    for (int tn = 0; tn < 8; ++tn)
      attw[(size_t)b * (N_DIM * N_DIM) + (size_t)row * N_DIM + tn * 16 + l15] =
          (bf16)(vv[tn] * inv);
  }
}

// ---------------------------------------------------------------------------
// LayerNorm over D=512 per row, one wave per row.
__global__ __launch_bounds__(256)
void ln_kernel(const bf16* __restrict__ h, const float* __restrict__ g,
               const float* __restrict__ bb, bf16* __restrict__ out) {
  const int row = (blockIdx.x << 3) + (threadIdx.x >> 5);
  const int lane = threadIdx.x & 31;
  const bf16* p = h + (size_t)row * D_DIM;
  float v[16];
  float s = 0.f;
#pragma unroll
  for (int t = 0; t < 16; ++t) { v[t] = (float)p[lane + 32 * t]; s += v[t]; }
  s = redsum32(s);
  const float mu = s * (1.f / D_DIM);
  float var = 0.f;
#pragma unroll
  for (int t = 0; t < 16; ++t) { const float d = v[t] - mu; var += d * d; }
  var = redsum32(var) * (1.f / D_DIM);
  const float inv = rsqrtf(var + 1e-5f);
#pragma unroll
  for (int t = 0; t < 16; ++t) {
    const int d0 = lane + 32 * t;
    out[(size_t)row * D_DIM + d0] = (bf16)((v[t] - mu) * inv * g[d0] + bb[d0]);
  }
}

// ---------------------------------------------------------------------------
extern "C" void kernel_launch(void* const* d_in, const int* in_sizes, int n_in,
                              void* d_out, int out_size, void* d_ws,
                              size_t ws_size, hipStream_t stream) {
  (void)in_sizes; (void)n_in; (void)out_size; (void)ws_size;
  const float* x    = (const float*)d_in[0];
  const float* wts  = (const float*)d_in[1];
  const float* embW = (const float*)d_in[2];
  const float* embB = (const float*)d_in[3];
  const float* wqW  = (const float*)d_in[4];
  const float* wqB  = (const float*)d_in[5];
  const float* wkW  = (const float*)d_in[6];
  const float* wkB  = (const float*)d_in[7];
  const float* wvW  = (const float*)d_in[8];
  const float* wvB  = (const float*)d_in[9];
  const float* ipW  = (const float*)d_in[10];
  const float* ipB  = (const float*)d_in[11];
  const float* lnG  = (const float*)d_in[12];
  const float* lnB  = (const float*)d_in[13];
  const float* outW = (const float*)d_in[14];
  const float* outB = (const float*)d_in[15];
  float* out = (float*)d_out;

  char* base = (char*)d_ws;
  size_t off = 0;
  auto alloc = [&](size_t bytes) -> char* {
    off = (off + 255) & ~(size_t)255;
    char* p = base + off;
    off += bytes;
    return p;
  };
  const size_t BN = (size_t)B_DIM * N_DIM;          // 65536 rows
  const size_t DD = (size_t)D_DIM * D_DIM;

  bf16* wembB = (bf16*)alloc((size_t)D_DIM * E_DIM * 2);
  bf16* wqC   = (bf16*)alloc(DD * 2);
  bf16* wkC   = (bf16*)alloc(DD * 2);
  bf16* wvC   = (bf16*)alloc(DD * 2);
  bf16* wiqC  = (bf16*)alloc(DD * 2);
  bf16* wikC  = (bf16*)alloc(DD * 2);
  bf16* woutC = (bf16*)alloc((size_t)E_DIM * D_DIM * 2);
  bf16* xbf   = (bf16*)alloc(BN * E_DIM * 2);
  bf16* xe    = (bf16*)alloc(BN * D_DIM * 2);
  float* padf = (float*)alloc(BN * 4);
  bf16* qin   = (bf16*)alloc(BN * D_DIM * 2);
  bf16* kin   = (bf16*)alloc(BN * D_DIM * 2);
  bf16* vinT  = (bf16*)alloc(BN * D_DIM * 2);       // per batch: D x N
  bf16* qh    = (bf16*)alloc(BN * D_DIM * 2);
  bf16* kh    = (bf16*)alloc(BN * D_DIM * 2);
  bf16* attw  = (bf16*)alloc(BN * N_DIM * 2);
  bf16* hpre  = qin;                                // reuse (dead after qh)
  bf16* hln   = kin;                                // reuse (dead after kh)

  auto cvt = [&](const float* s, bf16* d, size_t n) {
    cvt_f32_bf16<<<dim3(2048), dim3(256), 0, stream>>>(s, d, (long)n);
  };
  auto gemm = [&](const bf16* A, const bf16* W, const float* bias,
                  const bf16* resid, bf16* Yb, float* Yf, int M, int N, int K,
                  long wbs, int mode) {
    dim3 grid(N / 128, M / 64);
    gemm_bf16<<<grid, 256, (size_t)K * 128, stream>>>(A, W, bias, resid, Yb,
                                                      Yf, M, N, K, wbs, mode);
  };

  // ---- stage 0: bf16 conversions --------------------------------------
  cvt(x, xbf, BN * E_DIM);
  cvt(embW, wembB, (size_t)D_DIM * E_DIM);
  cvt(wqW, wqC, DD);
  cvt(wkW, wkC, DD);
  cvt(wvW, wvC, DD);
  cvt(ipW, wiqC, DD);                               // in_proj rows [0, D)
  cvt(ipW + DD, wikC, DD);                          // in_proj rows [D, 2D)
  cvt(outW, woutC, (size_t)E_DIM * D_DIM);

  // ---- stage 1: xe = x @ emb_W^T + b; pad mask ------------------------
  gemm(xbf, wembB, embB, nullptr, xe, nullptr, (int)BN, D_DIM, E_DIM, 0, 0);
  pad_kernel<<<dim3(BN / 8), dim3(256), 0, stream>>>(xe, padf);

  // ---- stage 2: q_in/k_in/v_in projections (v stored transposed) ------
  gemm(xe, wqC, wqB, nullptr, qin, nullptr, (int)BN, D_DIM, D_DIM, 0, 0);
  gemm(xe, wkC, wkB, nullptr, kin, nullptr, (int)BN, D_DIM, D_DIM, 0, 0);
  gemm(xe, wvC, wvB, nullptr, vinT, nullptr, (int)BN, D_DIM, D_DIM, 0, 1);

  // ---- stage 3: q/k head projections ----------------------------------
  gemm(qin, wiqC, ipB, nullptr, qh, nullptr, (int)BN, D_DIM, D_DIM, 0, 0);
  gemm(kin, wikC, ipB + D_DIM, nullptr, kh, nullptr, (int)BN, D_DIM, D_DIM, 0, 0);

  // ---- stage 4: fused attention weights -------------------------------
  attn_fused<<<dim3(B_DIM), dim3(256), 0, stream>>>(qh, kh, padf, wts, attw);

  // ---- stage 5: att_out = attw @ v_in; h = xe + softplus(att_out) -----
  gemm(attw, vinT, nullptr, xe, hpre, nullptr, (int)BN, D_DIM, N_DIM,
       (long)D_DIM * N_DIM, 2);

  // ---- stage 6: LayerNorm ---------------------------------------------
  ln_kernel<<<dim3(BN / 8), dim3(256), 0, stream>>>(hpre, lnG, lnB, hln);

  // ---- stage 7: out = h @ out_W^T + b (f32) ---------------------------
  gemm(hln, woutC, outB, nullptr, nullptr, out, (int)BN, E_DIM, D_DIM, 0, 3);
}
